// MinGRUParallelLayer_11957188952156
// MI455X (gfx1250) — matrix operand order repaired and checked by reading, not compiled, not run on hardware
//
#include <hip/hip_runtime.h>
#include <hip/hip_bf16.h>

#define T_STEPS 16384
#define D       512
#define CHUNKS  256
#define CLEN    64      // CHUNKS * CLEN == T_STEPS

typedef __attribute__((ext_vector_type(16))) __bf16 v16bf;
typedef __attribute__((ext_vector_type(8)))  __bf16 v8bf;
typedef __attribute__((ext_vector_type(8)))  float  v8f;

// ---------------------------------------------------------------------------
// f32 -> bf16 conversion (grid-stride)
// ---------------------------------------------------------------------------
__global__ __launch_bounds__(256)
void cvt_f32_bf16(const float* __restrict__ in, __bf16* __restrict__ out, int n) {
  for (int i = blockIdx.x * blockDim.x + threadIdx.x; i < n;
       i += gridDim.x * blockDim.x)
    out[i] = (__bf16)in[i];
}

// ---------------------------------------------------------------------------
// WMMA fragment layout (wave32, v_wmma_f32_16x16x32_bf16):
//  A (16x32 MxK): lane m (0..15)/m+16: elems 0..7 = K h*16..+7,
//                 elems 8..15 = K h*16+8..+15  (h = lane>>4, row = lane&15)
//  B (32x16 KxN): lane h*16+n holds column n, elems e = K k0+h*16+e
//  C/D (16x16):   elem r <-> (M = r + 8*h, N = lane&15)
// ---------------------------------------------------------------------------
union AFrag { v16bf v; v8bf p[2]; };

__device__ __forceinline__ v16bf load_a_frag(const __bf16* __restrict__ row,
                                             int k0, int h) {
  AFrag f;
  f.p[0] = *(const v8bf*)(row + k0 + h * 16);
  f.p[1] = *(const v8bf*)(row + k0 + h * 16 + 8);
  return f.v;
}

#define WMMA_BF16(A, B, C)                                                  \
  __builtin_amdgcn_wmma_f32_16x16x32_bf16(false, (A), false, (B), (short)0, \
                                          (C), false, false)

// ---------------------------------------------------------------------------
// Kernel 1: dual GEMM z = x@Wz^T, hcand = x@Wh^T, fused gate epilogue:
//   a = 1 - sigmoid(z + bz),  b = sigmoid(z + bz) * (hcand + bh)
// Wave tile: 64 rows x 16 cols for BOTH matrices (B frag reused by 4 WMMAs).
// Explicit ping-pong double buffering: step k+1 fragment loads are issued
// before step k's WMMAs so the s_wait before WMMAs leaves them in flight.
// Block: 8 waves = 64 rows x 128 cols.  Grid: (T/64, D/128).
// ---------------------------------------------------------------------------
__global__ __launch_bounds__(256)
void gemm_zh_kernel(const __bf16* __restrict__ X,
                    const __bf16* __restrict__ Wz,
                    const __bf16* __restrict__ Wh,
                    const float*  __restrict__ bzv,
                    const float*  __restrict__ bhv,
                    float* __restrict__ aBuf,
                    float* __restrict__ bBuf) {
  const int wave = threadIdx.x >> 5;
  const int lane = threadIdx.x & 31;
  const int m    = lane & 15;
  const int h    = lane >> 4;
  const int m0   = blockIdx.x * 64;
  const int col  = blockIdx.y * 128 + wave * 16 + m;  // output col == W row

  v8f accz[4] = {};
  v8f acch[4] = {};

  const __bf16* xr0 = X + (size_t)(m0 + m) * D;
  const __bf16* wzr = Wz + (size_t)col * D;
  const __bf16* whr = Wh + (size_t)col * D;

  v16bf Af[2][4];
  v16bf Bzf[2], Bhf[2];

  // prologue: stage k-step 0
  Bzf[0] = *(const v16bf*)(wzr + h * 16);
  Bhf[0] = *(const v16bf*)(whr + h * 16);
#pragma unroll
  for (int mt = 0; mt < 4; ++mt)
    Af[0][mt] = load_a_frag(xr0 + (size_t)(mt * 16) * D, 0, h);

#pragma unroll
  for (int ks = 0; ks < 16; ++ks) {
    const int cur = ks & 1;
    const int nxt = cur ^ 1;
    if (ks < 15) {                      // issue step k+1 loads first
      const int kn = (ks + 1) * 32;
      Bzf[nxt] = *(const v16bf*)(wzr + kn + h * 16);
      Bhf[nxt] = *(const v16bf*)(whr + kn + h * 16);
#pragma unroll
      for (int mt = 0; mt < 4; ++mt)
        Af[nxt][mt] = load_a_frag(xr0 + (size_t)(mt * 16) * D, kn, h);
    }
#pragma unroll
    for (int mt = 0; mt < 4; ++mt)
      accz[mt] = WMMA_BF16(Af[cur][mt], Bzf[cur], accz[mt]);
#pragma unroll
    for (int mt = 0; mt < 4; ++mt)
      acch[mt] = WMMA_BF16(Af[cur][mt], Bhf[cur], acch[mt]);
  }

  const float biasz = bzv[col];
  const float biash = bhv[col];
#pragma unroll
  for (int mt = 0; mt < 4; ++mt) {
#pragma unroll
    for (int r = 0; r < 8; ++r) {
      const int row = m0 + mt * 16 + r + 8 * h;
      const float z  = 1.0f / (1.0f + __expf(-(accz[mt][r] + biasz)));
      const float hc = acch[mt][r] + biash;
      aBuf[(size_t)row * D + col] = 1.0f - z;
      bBuf[(size_t)row * D + col] = z * hc;
    }
  }
}

// ---------------------------------------------------------------------------
// Scan pass 1: per-chunk aggregate (A = prod a, B = scan of b)
// ---------------------------------------------------------------------------
__global__ __launch_bounds__(512)
void scan_agg_kernel(const float* __restrict__ aB, const float* __restrict__ bB,
                     float* __restrict__ Aagg, float* __restrict__ Bagg) {
  const int ch = threadIdx.x;
  const int c  = blockIdx.x;
  float A = 1.0f, B = 0.0f;
  const int base = c * CLEN;
  for (int i = 0; i < CLEN; ++i) {
    const size_t idx = (size_t)(base + i) * D + ch;
    const float a = aB[idx];
    B = a * B + bB[idx];
    A *= a;
  }
  Aagg[c * D + ch] = A;
  Bagg[c * D + ch] = B;
}

// ---------------------------------------------------------------------------
// Scan pass 2: exclusive carry scan across chunks (one block, 512 channels)
// ---------------------------------------------------------------------------
__global__ __launch_bounds__(512)
void scan_carry_kernel(const float* __restrict__ Aagg,
                       const float* __restrict__ Bagg,
                       float* __restrict__ carry) {
  const int ch = threadIdx.x;
  float P = 0.0f;  // h_0 = 0
  for (int c = 0; c < CHUNKS; ++c) {
    carry[c * D + ch] = P;
    P = Aagg[c * D + ch] * P + Bagg[c * D + ch];
  }
}

// ---------------------------------------------------------------------------
// Scan pass 3: apply carry, emit states as bf16 (input to output GEMM)
// ---------------------------------------------------------------------------
__global__ __launch_bounds__(512)
void scan_apply_kernel(const float* __restrict__ aB, const float* __restrict__ bB,
                       const float* __restrict__ carry,
                       __bf16* __restrict__ Sbf) {
  const int ch = threadIdx.x;
  const int c  = blockIdx.x;
  float s = carry[c * D + ch];
  const int base = c * CLEN;
  for (int i = 0; i < CLEN; ++i) {
    const size_t idx = (size_t)(base + i) * D + ch;
    s = aB[idx] * s + bB[idx];
    Sbf[idx] = (__bf16)s;
  }
}

// ---------------------------------------------------------------------------
// Kernel 4: output GEMM  out = states @ Wo^T + bo  (f32 result)
// Wave tile: 64 rows x 32 cols (2 B frags reused by 4 WMMAs each), same
// explicit ping-pong pipeline.  Block: 8 waves = 64 rows x 256 cols.
// Grid: (T/64, D/256).
// ---------------------------------------------------------------------------
__global__ __launch_bounds__(256)
void gemm_out_kernel(const __bf16* __restrict__ S,
                     const __bf16* __restrict__ Wo,
                     const float*  __restrict__ bov,
                     float* __restrict__ out) {
  const int wave = threadIdx.x >> 5;
  const int lane = threadIdx.x & 31;
  const int m    = lane & 15;
  const int h    = lane >> 4;
  const int m0   = blockIdx.x * 64;
  const int c0   = blockIdx.y * 256 + wave * 32 + m;  // nt=0 col; nt=1 is +16

  v8f acc[2][4] = {};

  const __bf16* sr0 = S + (size_t)(m0 + m) * D;
  const __bf16* wr0 = Wo + (size_t)c0 * D;
  const __bf16* wr1 = Wo + (size_t)(c0 + 16) * D;

  v16bf Af[2][4];
  v16bf B0f[2], B1f[2];

  B0f[0] = *(const v16bf*)(wr0 + h * 16);
  B1f[0] = *(const v16bf*)(wr1 + h * 16);
#pragma unroll
  for (int mt = 0; mt < 4; ++mt)
    Af[0][mt] = load_a_frag(sr0 + (size_t)(mt * 16) * D, 0, h);

#pragma unroll
  for (int ks = 0; ks < 16; ++ks) {
    const int cur = ks & 1;
    const int nxt = cur ^ 1;
    if (ks < 15) {
      const int kn = (ks + 1) * 32;
      B0f[nxt] = *(const v16bf*)(wr0 + kn + h * 16);
      B1f[nxt] = *(const v16bf*)(wr1 + kn + h * 16);
#pragma unroll
      for (int mt = 0; mt < 4; ++mt)
        Af[nxt][mt] = load_a_frag(sr0 + (size_t)(mt * 16) * D, kn, h);
    }
#pragma unroll
    for (int mt = 0; mt < 4; ++mt)
      acc[0][mt] = WMMA_BF16(Af[cur][mt], B0f[cur], acc[0][mt]);
#pragma unroll
    for (int mt = 0; mt < 4; ++mt)
      acc[1][mt] = WMMA_BF16(Af[cur][mt], B1f[cur], acc[1][mt]);
  }

#pragma unroll
  for (int nt = 0; nt < 2; ++nt) {
    const int col = c0 + nt * 16;
    const float bias = bov[col];
#pragma unroll
    for (int mt = 0; mt < 4; ++mt) {
#pragma unroll
      for (int r = 0; r < 8; ++r) {
        const int row = m0 + mt * 16 + r + 8 * h;
        out[(size_t)row * D + col] = acc[nt][mt][r] + bias;
      }
    }
  }
}

// ---------------------------------------------------------------------------
// Launch
// ---------------------------------------------------------------------------
extern "C" void kernel_launch(void* const* d_in, const int* in_sizes, int n_in,
                              void* d_out, int out_size, void* d_ws, size_t ws_size,
                              hipStream_t stream) {
  const float* xs = (const float*)d_in[0];
  const float* Wz = (const float*)d_in[1];
  const float* bz = (const float*)d_in[2];
  const float* Wh = (const float*)d_in[3];
  const float* bh = (const float*)d_in[4];
  const float* Wo = (const float*)d_in[5];
  const float* bo = (const float*)d_in[6];
  float* out = (float*)d_out;

  // workspace layout (all 256-aligned by construction)
  char* ws = (char*)d_ws;
  const size_t SZ_XBF = (size_t)T_STEPS * D * sizeof(__bf16);   // 16 MB
  const size_t SZ_WBF = (size_t)D * D * sizeof(__bf16);         // 512 KB
  const size_t SZ_AB  = (size_t)T_STEPS * D * sizeof(float);    // 32 MB
  const size_t SZ_AGG = (size_t)CHUNKS * D * sizeof(float);     // 512 KB

  __bf16* xbf   = (__bf16*)(ws);                 size_t off = SZ_XBF;
  __bf16* wzbf  = (__bf16*)(ws + off);           off += SZ_WBF;
  __bf16* whbf  = (__bf16*)(ws + off);           off += SZ_WBF;
  __bf16* wobf  = (__bf16*)(ws + off);           off += SZ_WBF;
  float*  aBuf  = (float*) (ws + off);           off += SZ_AB;
  float*  bBuf  = (float*) (ws + off);           off += SZ_AB;
  float*  Aagg  = (float*) (ws + off);           off += SZ_AGG;
  float*  Bagg  = (float*) (ws + off);           off += SZ_AGG;
  float*  carry = (float*) (ws + off);           off += SZ_AGG;
  __bf16* Sbf   = (__bf16*)(ws + off);           off += SZ_XBF;

  const int NX = T_STEPS * D;
  const int NW = D * D;

  cvt_f32_bf16<<<(NX + 255) / 256, 256, 0, stream>>>(xs, xbf, NX);
  cvt_f32_bf16<<<(NW + 255) / 256, 256, 0, stream>>>(Wz, wzbf, NW);
  cvt_f32_bf16<<<(NW + 255) / 256, 256, 0, stream>>>(Wh, whbf, NW);
  cvt_f32_bf16<<<(NW + 255) / 256, 256, 0, stream>>>(Wo, wobf, NW);

  gemm_zh_kernel<<<dim3(T_STEPS / 64, D / 128), 256, 0, stream>>>(
      xbf, wzbf, whbf, bz, bh, aBuf, bBuf);

  scan_agg_kernel  <<<CHUNKS, D, 0, stream>>>(aBuf, bBuf, Aagg, Bagg);
  scan_carry_kernel<<<1,      D, 0, stream>>>(Aagg, Bagg, carry);
  scan_apply_kernel<<<CHUNKS, D, 0, stream>>>(aBuf, bBuf, carry, Sbf);

  gemm_out_kernel<<<dim3(T_STEPS / 64, D / 256), 256, 0, stream>>>(
      Sbf, wobf, bo, out);
}
